// CrissCrossAttention_4028679324314
// MI455X (gfx1250) — compile-verified, matrix operand order checked by reading, not run on hardware
//
#include <hip/hip_runtime.h>
#include <hip/hip_bf16.h>

typedef __bf16 bf16_t;
typedef __attribute__((ext_vector_type(16))) __bf16 v16bf;
typedef __attribute__((ext_vector_type(8)))  __bf16 bf16x8;
typedef __attribute__((ext_vector_type(2)))  __bf16 bf16x2;
typedef __attribute__((ext_vector_type(8)))  float  v8f;

constexpr int B  = 8;
constexpr int C  = 64;
constexpr int CQ = 8;
constexpr int H  = 256;
constexpr int W  = 256;
constexpr int HW = H * W;   // 65536

// Concatenate two contiguous 8x-bf16 chunks (2x ds_load_b128) into a WMMA
// 16-bit fragment. Per CDNA5 ISA 7.12.2, a lane's v16bf holds K-runs
// [base..base+7] then [base+16..base+23], base = (lane>=16) ? 8 : 0.
__device__ __forceinline__ v16bf load_frag16(const bf16_t* p) {
  bf16x8 lo = *(const bf16x8*)p;        // K = base .. base+7
  bf16x8 hi = *(const bf16x8*)(p + 16); // K = base+16 .. base+23
  return __builtin_shufflevector(lo, hi, 0, 1, 2, 3, 4, 5, 6, 7,
                                 8, 9, 10, 11, 12, 13, 14, 15);
}

// ---------------------------------------------------------------------------
// Kernel 1: q/k 1x1-conv projections (memory-bound, coalesced along W)
// ---------------------------------------------------------------------------
__global__ __launch_bounds__(256)
void cca_qk_kernel(const float* __restrict__ x,
                   const float* __restrict__ Wq, const float* __restrict__ bq,
                   const float* __restrict__ Wk, const float* __restrict__ bk,
                   float* __restrict__ qws, float* __restrict__ kws)
{
  __shared__ float Wqs[CQ][C];
  __shared__ float Wks[CQ][C];
  __shared__ float bqs[CQ];
  __shared__ float bks[CQ];

  const int tid = threadIdx.x;
  for (int idx = tid; idx < CQ * C; idx += 256) {
    Wqs[idx >> 6][idx & 63] = Wq[idx];
    Wks[idx >> 6][idx & 63] = Wk[idx];
  }
  if (tid < CQ) { bqs[tid] = bq[tid]; bks[tid] = bk[tid]; }
  __syncthreads();

  const size_t p   = (size_t)blockIdx.x * 256 + tid;   // pixel id over B*HW
  const int    b   = (int)(p >> 16);                   // HW == 65536
  const int    rem = (int)(p & 65535);                 // h*W + w

  float aq[CQ], ak[CQ];
#pragma unroll
  for (int o = 0; o < CQ; ++o) { aq[o] = 0.f; ak[o] = 0.f; }

  const size_t xb = (size_t)b * C * HW + rem;
  for (int c = 0; c < C; ++c) {
    float xv = x[xb + (size_t)c * HW];
#pragma unroll
    for (int o = 0; o < CQ; ++o) {
      aq[o] = fmaf(Wqs[o][c], xv, aq[o]);
      ak[o] = fmaf(Wks[o][c], xv, ak[o]);
    }
  }
  const size_t qb = (size_t)b * CQ * HW + rem;
#pragma unroll
  for (int o = 0; o < CQ; ++o) {
    qws[qb + (size_t)o * HW] = aq[o] + bqs[o];
    kws[qb + (size_t)o * HW] = ak[o] + bks[o];
  }
}

// ---------------------------------------------------------------------------
// Kernels 2/3 share this body. AXIS_COL: one WG per (b,w), attention along H.
// AXIS_ROW: one WG per (b,h), attention along W; fuses gamma*(h+w)+x output.
// ---------------------------------------------------------------------------
template <bool AXIS_COL>
__global__ __launch_bounds__(256)
void cca_attn_kernel(const float* __restrict__ x,
                     const float* __restrict__ qws, const float* __restrict__ kws,
                     const float* __restrict__ Wv,  const float* __restrict__ bv,
                     const float* __restrict__ gamma,
                     float* __restrict__ out)
{
  __shared__ float  Qs[CQ][256];                       //  8 KB
  __shared__ float  Ks[CQ][256];                       //  8 KB
  __shared__ __align__(16) bf16_t Xsw[256][C];         // 32 KB  [pixel][chan] (K-major for B)
  __shared__ __align__(16) bf16_t Vs[C][256];          // 32 KB  [chan][pixel] (K-major for A)
  __shared__ __align__(16) bf16_t Wvs[C][C];           //  8 KB  row-major (K-major for A)
  __shared__ __align__(16) bf16_t Psw[2][256][32];     // 2x16 KB double-buffered P slab
  __shared__ float  bvs[C];
  __shared__ float  m_s[256];
  __shared__ float  r_s[256];

  const int tid  = threadIdx.x;
  const int lane = tid & 31;
  const int wave = tid >> 5;
  const int lb   = (lane >= 16) ? 8 : 0;   // per-lane K base
  const int l15  = lane & 15;

  const int blk = blockIdx.x;          // b*W + w (col)  or  b*H + h (row)
  const int b   = blk >> 8;
  const int pos = blk & 255;           // w (col) or h (row)
  const size_t xbase = (size_t)b * C * HW;
  const size_t lstride = AXIS_COL ? (size_t)W : (size_t)1;
  const size_t loff    = AXIS_COL ? (size_t)pos : (size_t)pos * W;

  // ---- stage Wv (bf16), bv, Q/K line, X line (transposed, pair-packed) ------
  for (int idx = tid; idx < C * C; idx += 256)
    Wvs[idx >> 6][idx & 63] = (bf16_t)Wv[idx];
  if (tid < C) bvs[tid] = bv[tid];

  const size_t qb = (size_t)b * CQ * HW + loff;
  for (int idx = tid; idx < CQ * 256; idx += 256) {
    int c = idx >> 8, i = idx & 255;
    Qs[c][i] = qws[qb + (size_t)c * HW + (size_t)i * lstride];
    Ks[c][i] = kws[qb + (size_t)c * HW + (size_t)i * lstride];
  }
  for (int idx = tid; idx < (C / 2) * 256; idx += 256) {
    int cp = idx >> 8;                 // channel pair 0..31
    int i  = idx & 255;                // pixel along the attended line
    const size_t base = xbase + loff + (size_t)i * lstride;
    bf16x2 pr;
    pr[0] = (bf16_t)x[base + (size_t)(2 * cp)     * HW];
    pr[1] = (bf16_t)x[base + (size_t)(2 * cp + 1) * HW];
    *(bf16x2*)&Xsw[i][2 * cp] = pr;
  }
  __syncthreads();

  // ---- V = Wv * X + bv  (bf16 WMMA, f32 accumulate) -------------------------
  // Each wave: fixed mb = wave>>1, 8 n-tiles. A fragment shared across tiles.
  {
    const int mb  = wave >> 1;
    const int nb0 = (wave & 1) << 3;
    const int m   = mb * 16 + l15;
    v8f acc[8];
    const v8f vzero = {0.f, 0.f, 0.f, 0.f, 0.f, 0.f, 0.f, 0.f};
#pragma unroll
    for (int t = 0; t < 8; ++t) acc[t] = vzero;

#pragma unroll
    for (int ks = 0; ks < 2; ++ks) {
      const int k0 = ks * 32;
      const v16bf a = load_frag16(&Wvs[m][k0 + lb]);   // 2x ds_load_b128
#pragma unroll
      for (int t = 0; t < 8; ++t) {
        const int n = (nb0 + t) * 16 + l15;
        const v16bf bm = load_frag16(&Xsw[n][k0 + lb]);
        acc[t] = __builtin_amdgcn_wmma_f32_16x16x32_bf16(
            false, a, false, bm, (short)0, acc[t], false, false);
      }
    }
#pragma unroll
    for (int t = 0; t < 8; ++t) {
      const int n = (nb0 + t) * 16 + l15;
#pragma unroll
      for (int v = 0; v < 8; ++v) {
        const int cR = mb * 16 + v + ((lane >> 4) << 3);
        Vs[cR][n] = (bf16_t)(acc[t][v] + bvs[cR]);
      }
    }
  }
  __syncthreads();

  // ---- softmax stats (f32): row i = tid of S, reduced over last axis --------
  {
    float qv[CQ];
#pragma unroll
    for (int c = 0; c < CQ; ++c) qv[c] = Qs[c][tid];
    float mval = -3.0e38f, zval = 0.0f;
    for (int j = 0; j < 256; ++j) {
      float s = 0.f;
#pragma unroll
      for (int c = 0; c < CQ; ++c) s = fmaf(qv[c], Ks[c][j], s);
      if (s > mval) { zval = zval * __expf(mval - s) + 1.0f; mval = s; }
      else          { zval += __expf(s - mval); }
    }
    m_s[tid] = mval;
    r_s[tid] = 1.0f / zval;
  }
  __syncthreads();

  // ---- out_line = V(64x256) * P(256x256), K-slabs of 32, double-buffered ----
  const int mb  = wave >> 1;
  const int nb0 = (wave & 1) << 3;
  const int m   = mb * 16 + l15;

  v8f acc[8];
  const v8f vzero = {0.f, 0.f, 0.f, 0.f, 0.f, 0.f, 0.f, 0.f};
#pragma unroll
  for (int t = 0; t < 8; ++t) acc[t] = vzero;

  float kv[CQ];                         // K column for j = tid, in registers
#pragma unroll
  for (int c = 0; c < CQ; ++c) kv[c] = Ks[c][tid];

  // build one K-major bf16 P slab (thread owns column j = tid, 32 slab rows,
  // written as 4x ds_store_b128)
  auto build_p = [&](int ib, int buf) {
    const int i0 = ib * 32;
#pragma unroll
    for (int e0 = 0; e0 < 32; e0 += 8) {
      bf16x8 pk;
#pragma unroll
      for (int e = 0; e < 8; ++e) {
        const int i = i0 + e0 + e;
        float s = 0.f;
#pragma unroll
        for (int c = 0; c < CQ; ++c) s = fmaf(Qs[c][i], kv[c], s);
        pk[e] = (bf16_t)(__expf(s - m_s[i]) * r_s[i]);
      }
      *(bf16x8*)&Psw[buf][tid][e0] = pk;
    }
  };

  build_p(0, 0);
  __syncthreads();

  for (int ib = 0; ib < 8; ++ib) {
    const int cur = ib & 1;
    // software pipeline: VALU build of the next slab co-executes with the
    // XDL WMMA stream consuming the current slab (other waves overlap too).
    if (ib < 7) build_p(ib + 1, cur ^ 1);

    const v16bf a = load_frag16(&Vs[m][ib * 32 + lb]); // 2x ds_load_b128
#pragma unroll
    for (int t = 0; t < 8; ++t) {
      const int n = (nb0 + t) * 16 + l15;
      const v16bf bm = load_frag16(&Psw[cur][n][lb]);
      acc[t] = __builtin_amdgcn_wmma_f32_16x16x32_bf16(
          false, a, false, bm, (short)0, acc[t], false, false);
    }
    __syncthreads();   // slab ib consumed by all waves; buffer reusable
  }

  // ---- writeback ------------------------------------------------------------
  const float g = *gamma;
#pragma unroll
  for (int t = 0; t < 8; ++t) {
    const int n = (nb0 + t) * 16 + l15;
#pragma unroll
    for (int v = 0; v < 8; ++v) {
      const int cR = mb * 16 + v + ((lane >> 4) << 3);
      const size_t idx = xbase + (size_t)cR * HW + loff + (size_t)n * lstride;
      if (AXIS_COL) {
        out[idx] = acc[t][v];                             // raw h_out (pass 1)
      } else {
        out[idx] = g * (out[idx] + acc[t][v]) + x[idx];   // fused final
      }
    }
  }
}

// ---------------------------------------------------------------------------
extern "C" void kernel_launch(void* const* d_in, const int* in_sizes, int n_in,
                              void* d_out, int out_size, void* d_ws, size_t ws_size,
                              hipStream_t stream) {
  const float* x     = (const float*)d_in[0];
  const float* Wq    = (const float*)d_in[1];
  const float* bq    = (const float*)d_in[2];
  const float* Wk    = (const float*)d_in[3];
  const float* bk    = (const float*)d_in[4];
  const float* Wv    = (const float*)d_in[5];
  const float* bv    = (const float*)d_in[6];
  const float* gamma = (const float*)d_in[7];
  float* out = (float*)d_out;

  float* qws = (float*)d_ws;                       // B*CQ*HW floats (16 MB)
  float* kws = qws + (size_t)B * CQ * HW;          // +16 MB

  cca_qk_kernel<<<(B * HW) / 256, 256, 0, stream>>>(x, Wq, bq, Wk, bk, qws, kws);
  cca_attn_kernel<true ><<<B * W, 256, 0, stream>>>(x, qws, kws, Wv, bv, gamma, out);
  cca_attn_kernel<false><<<B * H, 256, 0, stream>>>(x, qws, kws, Wv, bv, gamma, out);
}
